// GNNLayer_86887188398821
// MI455X (gfx1250) — matrix operand order repaired and checked by reading, not compiled, not run on hardware
//
#include <hip/hip_runtime.h>
#include <hip/hip_bf16.h>
#include <cstdint>

// GNN layer: out = relu(x @ W^T + b); then out[col[e],0] += x[e,0]
// GEMM via 3-term bf16-split WMMA (fp32-class accuracy at bf16 rate).
// Block tile 128x128, 8 waves (4m x 2n), each wave 32x64 -> 24 WMMA / K-step.

typedef __attribute__((ext_vector_type(16))) __bf16 v16bf;
typedef __attribute__((ext_vector_type(8)))  float  v8f;

#define KDIM 512
#define NOUT 512
#define BM   128
#define BN   128
#define KC   32
#define LDP  (KC + 4)   // 36 shorts = 72B row stride: 8B-aligned rows, conflict-free frag reads

union FragBF { v16bf v; uint32_t u[8]; };
union AccF   { v8f   v; float    f[8]; };

__device__ __forceinline__ float bf16_hi_part(float a) {
    return __uint_as_float(__float_as_uint(a) & 0xffff0000u);
}
__device__ __forceinline__ uint32_t pack_hi(float a, float b) {
    return (__float_as_uint(a) >> 16) | (__float_as_uint(b) & 0xffff0000u);
}
__device__ __forceinline__ uint32_t pack_lo(float a, float b) {
    float ra = a - bf16_hi_part(a);          // exact residual of bf16 truncation
    float rb = b - bf16_hi_part(b);
    return (__float_as_uint(ra) >> 16) | (__float_as_uint(rb) & 0xffff0000u);
}

__launch_bounds__(256)
__global__ void gemm_bias_relu_bf16x3(const float* __restrict__ x,
                                      const float* __restrict__ W,
                                      const float* __restrict__ bias,
                                      float* __restrict__ out,
                                      int nrows)
{
    __shared__ uint16_t Axh[BM][LDP];
    __shared__ uint16_t Axl[BM][LDP];
    __shared__ uint16_t Bwh[BN][LDP];
    __shared__ uint16_t Bwl[BN][LDP];

    const int tid  = threadIdx.x;
    const int lane = tid & 31;          // wave32
    const int wave = tid >> 5;          // 8 waves: 4(m) x 2(n)
    const int bm0  = blockIdx.y * BM;
    const int bn0  = blockIdx.x * BN;
    const int m0   = (wave >> 1) * 32;  // 32 rows per wave
    const int n0   = (wave & 1) * 64;   // 64 cols per wave

    // staging: both tiles are 128x32 f32 = 1024 float4; 4 float4 per thread
    const int r0   = tid >> 3;          // staging row (+ i*32)
    const int scol = (tid & 7) * 4;     // staging col (float index)

    float4 xs[4], ws[4];

    auto load_stage = [&](int kk) {
        #pragma unroll
        for (int i = 0; i < 4; ++i) {
            int row = r0 + i * 32;
            int gr  = bm0 + row;
            xs[i] = (gr < nrows)
                  ? *(const float4*)&x[(size_t)gr * KDIM + kk + scol]
                  : make_float4(0.f, 0.f, 0.f, 0.f);
            ws[i] = *(const float4*)&W[(size_t)(bn0 + row) * KDIM + kk + scol];
        }
    };
    auto store_stage = [&]() {
        #pragma unroll
        for (int i = 0; i < 4; ++i) {
            int row = r0 + i * 32;
            uint32_t* ph = (uint32_t*)&Axh[row][scol];
            uint32_t* pl = (uint32_t*)&Axl[row][scol];
            ph[0] = pack_hi(xs[i].x, xs[i].y); ph[1] = pack_hi(xs[i].z, xs[i].w);
            pl[0] = pack_lo(xs[i].x, xs[i].y); pl[1] = pack_lo(xs[i].z, xs[i].w);
            uint32_t* qh = (uint32_t*)&Bwh[row][scol];
            uint32_t* ql = (uint32_t*)&Bwl[row][scol];
            qh[0] = pack_hi(ws[i].x, ws[i].y); qh[1] = pack_hi(ws[i].z, ws[i].w);
            ql[0] = pack_lo(ws[i].x, ws[i].y); ql[1] = pack_lo(ws[i].z, ws[i].w);
        }
    };

    AccF acc[2][4];
    #pragma unroll
    for (int ms = 0; ms < 2; ++ms)
        #pragma unroll
        for (int nt = 0; nt < 4; ++nt)
            #pragma unroll
            for (int i = 0; i < 8; ++i) acc[ms][nt].f[i] = 0.0f;

    const int fr = lane & 15;   // fragment row/col within 16x16 tile
    const int fh = lane >> 4;   // K-half selector per ISA 7.12.2 layout

    load_stage(0);

    for (int kk = 0; kk < KDIM; kk += KC) {
        __syncthreads();                       // prior chunk's fragment reads done
        store_stage();                         // convert f32 -> {hi,lo} bf16 planes
        __syncthreads();

        if (kk + KC < KDIM) load_stage(kk + KC);   // overlap next chunk with compute
        {
            int pr = bm0 + r0;
            if (kk + 2 * KC < KDIM && pr < nrows)  // gfx1250 global_prefetch_b8
                __builtin_prefetch((const void*)&x[(size_t)pr * KDIM + kk + 2 * KC + scol], 0, 1);
        }

        // ---- A fragments (16x32 bf16 layout) for both m-subtiles ----
        FragBF AfH[2], AfL[2];
        #pragma unroll
        for (int ms = 0; ms < 2; ++ms) {
            int ar = m0 + ms * 16 + fr;
            #pragma unroll
            for (int v = 0; v < 8; ++v) {
                int kp = (v & 3) + ((v >> 2) << 3) + (fh << 2);  // K-pair index
                AfH[ms].u[v] = *(const uint32_t*)&Axh[ar][kp * 2];
                AfL[ms].u[v] = *(const uint32_t*)&Axl[ar][kp * 2];
            }
        }
        // ---- 4 n-subtiles: B frag amortized over 2 m-subtiles x 3 split terms ----
        #pragma unroll
        for (int nt = 0; nt < 4; ++nt) {
            FragBF BfH, BfL;
            int bc = n0 + nt * 16 + fr;        // B column = W-tile row
            #pragma unroll
            for (int v = 0; v < 8; ++v) {
                int kp = (v & 3) + ((v >> 2) << 3) + (fh << 2);
                BfH.u[v] = *(const uint32_t*)&Bwh[bc][kp * 2];
                BfL.u[v] = *(const uint32_t*)&Bwl[bc][kp * 2];
            }
            #pragma unroll
            for (int ms = 0; ms < 2; ++ms) {
                acc[ms][nt].v = __builtin_amdgcn_wmma_f32_16x16x32_bf16(
                    false, AfH[ms].v, false, BfH.v, (short)0, acc[ms][nt].v, false, false);
                acc[ms][nt].v = __builtin_amdgcn_wmma_f32_16x16x32_bf16(
                    false, AfH[ms].v, false, BfL.v, (short)0, acc[ms][nt].v, false, false);
                acc[ms][nt].v = __builtin_amdgcn_wmma_f32_16x16x32_bf16(
                    false, AfL[ms].v, false, BfH.v, (short)0, acc[ms][nt].v, false, false);
            }
        }
    }

    // ---- epilogue: bias + relu + store (C/D layout: VGPR r -> M=r+8h, lane -> N) ----
    #pragma unroll
    for (int ms = 0; ms < 2; ++ms) {
        #pragma unroll
        for (int nt = 0; nt < 4; ++nt) {
            int gcol = bn0 + n0 + nt * 16 + fr;
            float bv = bias[gcol];
            #pragma unroll
            for (int r = 0; r < 8; ++r) {
                int grow = bm0 + m0 + ms * 16 + r + 8 * fh;
                if (grow < nrows) {
                    float val = acc[ms][nt].f[r] + bv;
                    out[(size_t)grow * NOUT + gcol] = val > 0.0f ? val : 0.0f;
                }
            }
        }
    }
}

// out[col[e], 0] += x[e, 0]  (reference scatter_add touches only column 0)
__global__ void scatter_add_col0(const long long* __restrict__ ei,
                                 const float* __restrict__ x,
                                 float* __restrict__ out, int nE)
{
    int e = blockIdx.x * blockDim.x + threadIdx.x;
    if (e < nE) {
        long long c = ei[(size_t)nE + e];   // edge_index[1][e]
        atomicAdd(out + (size_t)c * NOUT, x[(size_t)e * KDIM]);
    }
}

extern "C" void kernel_launch(void* const* d_in, const int* in_sizes, int n_in,
                              void* d_out, int out_size, void* d_ws, size_t ws_size,
                              hipStream_t stream)
{
    const float*     x   = (const float*)d_in[0];
    const long long* ei  = (const long long*)d_in[1];
    const float*     W   = (const float*)d_in[2];
    const float*     b   = (const float*)d_in[3];
    float*           out = (float*)d_out;

    const int nrows = in_sizes[0] / KDIM;   // 100000
    const int nE    = in_sizes[1] / 2;      // 100000

    dim3 grid(NOUT / BN, (nrows + BM - 1) / BM);
    gemm_bias_relu_bf16x3<<<grid, 256, 0, stream>>>(x, W, b, out, nrows);
    scatter_add_col0<<<(nE + 255) / 256, 256, 0, stream>>>(ei, x, out, nE);
}
